// VQQuantizer_30064771072206
// MI455X (gfx1250) — matrix operand order repaired and checked by reading, not compiled, not run on hardware
//
#include <hip/hip_runtime.h>

// ---------------------------------------------------------------------------
// VQ quantizer for MI455X (gfx1250, wave32):
//   - bf16 WMMA 16x16x32 for the N x K x 64 distance matmul
//   - TDM (tensor_load_to_lds) double-buffered codebook staging
//   z: [16,1024,512] f32 -> N=131072 rows of D=64; codebook: [2048,64] f32
//   argmin_k ( ||c_k||^2 - 2 * z.c_k )   (||z||^2 constant per row, dropped)
// ---------------------------------------------------------------------------

typedef __attribute__((ext_vector_type(16))) __bf16 v16bf;
typedef __attribute__((ext_vector_type(8)))  float  v8f;
typedef __attribute__((ext_vector_type(4)))  unsigned int v4u;
typedef __attribute__((ext_vector_type(8)))  int  v8i;
typedef __attribute__((ext_vector_type(4)))  int  v4i;

#define NROWS          131072   // 16*1024*8 groups
#define DDIM           64
#define KCODES         2048
#define CHUNK          128      // codebook rows per LDS buffer (16 KB bf16)
#define NCH            (KCODES / CHUNK)
#define ROWS_PER_BLOCK 128
#define WAVES          8

#if defined(__has_builtin)
#  if __has_builtin(__builtin_amdgcn_tensor_load_to_lds) && \
      __has_builtin(__builtin_amdgcn_s_wait_tensorcnt)
#    define USE_TDM 1
#  endif
#endif
#if __has_include(<hip/amd_detail/amd_gfx1250_TDM.h>)
#  define TDM_6ARG 1        // therock-10.0 headers => 6-arg builtin form
#endif

__device__ __forceinline__ unsigned short f32_to_bf16(float f) {
  unsigned int u = __float_as_uint(f);
  u += 0x7FFFu + ((u >> 16) & 1u);          // round-to-nearest-even
  return (unsigned short)(u >> 16);
}

union BV { v16bf v; unsigned short u[16]; uint4 q[2]; };

#ifdef USE_TDM
// 1-D TDM copy: ndw dwords global -> LDS.  D# per CDNA5 ISA ch.8:
//   group0: count=1 | lds_addr | global_addr | type=2
//   group1: data_size=4B, tensor_dim0=tile_dim0=ndw, tensor_dim1=tile_dim1=1
__device__ __forceinline__ void tdm_load_1d(unsigned int ldsAddr,
                                            const void* gptr,
                                            unsigned int ndw) {
  unsigned long long ga = (unsigned long long)(uintptr_t)gptr;
  v4u g0;
  g0.x = 1u;                                      // count=1 (valid descriptor)
  g0.y = ldsAddr;                                 // lds_addr (bytes)
  g0.z = (unsigned int)ga;                        // global_addr[31:0]
  g0.w = (unsigned int)(ga >> 32) | (2u << 30);   // global_addr[56:32] | type=2
  v8i g1;
  g1[0] = (int)(2u << 16);                        // data_size=2 -> 4-byte elems
  g1[1] = (int)(ndw << 16);                       // tensor_dim0[15:0]
  g1[2] = (int)(1u << 16);                        // tensor_dim0 hi=0 | tensor_dim1=1
  g1[3] = (int)(ndw << 16);                       // tile_dim0
  g1[4] = 1;                                      // tile_dim1=1, tile_dim2=0
  g1[5] = (int)ndw;                               // tensor_dim0_stride lo
  g1[6] = 0;
  g1[7] = 0;
  v4i z4 = {0, 0, 0, 0};
#ifdef TDM_6ARG
  v8i z8 = {0, 0, 0, 0, 0, 0, 0, 0};
  __builtin_amdgcn_tensor_load_to_lds(g0, g1, z4, z4, z8, 0);
#else
  __builtin_amdgcn_tensor_load_to_lds(g0, g1, z4, z4, 0);
#endif
}
#endif

// --------------------------------------------------------------------------
// Kernel 1: codebook -> bf16 (workspace) + ||c||^2 (f32, workspace)
// --------------------------------------------------------------------------
__global__ void vq_prep(const float* __restrict__ cb,
                        unsigned short* __restrict__ cb16,
                        float* __restrict__ csq) {
  const int k = blockIdx.x * blockDim.x + threadIdx.x;
  if (k >= KCODES) return;
  const float* r = cb + (size_t)k * DDIM;
  float s = 0.f;
  #pragma unroll
  for (int d = 0; d < DDIM; ++d) {
    float x = r[d];
    s += x * x;
    cb16[(size_t)k * DDIM + d] = f32_to_bf16(x);
  }
  csq[k] = s;
}

// --------------------------------------------------------------------------
// Kernel 2: WMMA distance search + gather + per-block loss partials
// --------------------------------------------------------------------------
__launch_bounds__(256)
__global__ void vq_main(const float* __restrict__ z,
                        const float* __restrict__ cb,
                        const unsigned short* __restrict__ cb16,
                        const float* __restrict__ csq,
                        float* __restrict__ out_codes,
                        float* __restrict__ out_idx,
                        float* __restrict__ partials) {
  __shared__ unsigned short sCb[2][CHUNK * DDIM];   // 2 x 16 KB ping-pong
  __shared__ float sCsq[2][CHUNK];
  __shared__ int   sIdx[WAVES][16];
  __shared__ float sLoss[WAVES];

  const int tid  = threadIdx.x;
  const int wave = tid >> 5;
  const int lane = tid & 31;
  const int hi   = lane >> 4;
  const int lm   = lane & 15;
  const int rowBase = blockIdx.x * ROWS_PER_BLOCK + wave * 16;
  const int myRow   = rowBase + lm;

  // ---- A fragments (this lane's z row, bf16), register-resident all kernel.
  // ISA 16-bit A 16x32: elems 0..7 = K(hi*8..+7), elems 8..15 = K(16+hi*8..+7)
  BV a0, a1;
  {
    const float* zr = z + (size_t)myRow * DDIM;
    const int o = hi * 8;
    auto pack8 = [](float4 x, float4 y, unsigned short* d) {
      d[0]=f32_to_bf16(x.x); d[1]=f32_to_bf16(x.y); d[2]=f32_to_bf16(x.z); d[3]=f32_to_bf16(x.w);
      d[4]=f32_to_bf16(y.x); d[5]=f32_to_bf16(y.y); d[6]=f32_to_bf16(y.z); d[7]=f32_to_bf16(y.w);
    };
    const float4* p0 = (const float4*)(zr + o);
    const float4* p1 = (const float4*)(zr + 16 + o);
    const float4* p2 = (const float4*)(zr + 32 + o);
    const float4* p3 = (const float4*)(zr + 48 + o);
    pack8(p0[0], p0[1], &a0.u[0]);
    pack8(p1[0], p1[1], &a0.u[8]);
    pack8(p2[0], p2[1], &a1.u[0]);
    pack8(p3[0], p3[1], &a1.u[8]);
  }

  float best[8];
  int   bidx[8];
  #pragma unroll
  for (int r = 0; r < 8; ++r) { best[r] = 3.4e38f; bidx[r] = 0; }

#ifdef USE_TDM
  if (wave == 0)   // prime the pipeline: chunk 0 -> buffer 0 via TDM
    tdm_load_1d((unsigned int)(uintptr_t)&sCb[0][0], cb16, CHUNK * DDIM / 2);
#endif

  for (int ch = 0; ch < NCH; ++ch) {
    const int p = ch & 1;
    if (tid < CHUNK) sCsq[p][tid] = csq[ch * CHUNK + tid];
#ifdef USE_TDM
    if (wave == 0) {
      if (ch + 1 < NCH) {     // async-prefetch next chunk into the other buffer
        tdm_load_1d((unsigned int)(uintptr_t)&sCb[1 - p][0],
                    cb16 + (size_t)(ch + 1) * CHUNK * DDIM, CHUNK * DDIM / 2);
        __builtin_amdgcn_s_wait_tensorcnt(1);   // chunk ch landed; ch+1 in flight
      } else {
        __builtin_amdgcn_s_wait_tensorcnt(0);
      }
    }
#else
    {
      const unsigned int* src = (const unsigned int*)(cb16 + (size_t)ch * CHUNK * DDIM);
      unsigned int* dst = (unsigned int*)&sCb[p][0];
      #pragma unroll
      for (int i = 0; i < (CHUNK * DDIM / 2) / 256; ++i)
        dst[tid + i * 256] = src[tid + i * 256];
    }
#endif
    __syncthreads();

    #pragma unroll 2
    for (int sub = 0; sub < CHUNK; sub += 16) {
      const int n = sub + lm;
      // ISA 16-bit B 32x16: elems 0..7 = K(hi*8..+7), elems 8..15 = K(16+hi*8..+7)
      // -> two 16B LDS reads per fragment.
      BV b0, b1;
      const uint4* bp0 = (const uint4*)(&sCb[p][0] + n * DDIM + hi * 8);
      const uint4* bp1 = (const uint4*)(&sCb[p][0] + n * DDIM + 32 + hi * 8);
      b0.q[0] = bp0[0]; b0.q[1] = bp0[2];
      b1.q[0] = bp1[0]; b1.q[1] = bp1[2];

      v8f acc = {0.f,0.f,0.f,0.f,0.f,0.f,0.f,0.f};
      acc = __builtin_amdgcn_wmma_f32_16x16x32_bf16(false, a0.v, false, b0.v,
                                                    (short)0, acc, false, false);
      acc = __builtin_amdgcn_wmma_f32_16x16x32_bf16(false, a1.v, false, b1.v,
                                                    (short)0, acc, false, false);
      const float cs = sCsq[p][n];
      const int kabs = ch * CHUNK + n;
      #pragma unroll
      for (int r = 0; r < 8; ++r) {                // row = hi*8 + r
        float s = cs - 2.0f * acc[r];
        if (s < best[r]) { best[r] = s; bidx[r] = kabs; }
      }
    }
    __syncthreads();   // all waves done with buffer p before TDM overwrites it
  }

  // ---- argmin across the 16 code-lanes (xor masks < 16 keep hi halves)
  #pragma unroll
  for (int off = 1; off < 16; off <<= 1) {
    #pragma unroll
    for (int r = 0; r < 8; ++r) {
      float ov = __shfl_xor(best[r], off, 32);
      int   oi = __shfl_xor(bidx[r], off, 32);
      if (ov < best[r] || (ov == best[r] && oi < bidx[r])) {
        best[r] = ov; bidx[r] = oi;
      }
    }
  }
  if (lm == 0) {
    #pragma unroll
    for (int r = 0; r < 8; ++r) sIdx[wave][hi * 8 + r] = bidx[r];
  }
  __syncthreads();

  // ---- gather codes (fp32), indices, commit-loss partial
  float lossAcc = 0.f;
  #pragma unroll 1
  for (int r = 0; r < 16; ++r) {
    const int row = rowBase + r;
    const int k   = sIdx[wave][r];
    const float* cbr = cb + (size_t)k * DDIM;
    const float* zr  = z  + (size_t)row * DDIM;
    float c0 = cbr[lane], c1 = cbr[lane + 32];
    float z0 = zr[lane],  z1 = zr[lane + 32];
    out_codes[(size_t)row * DDIM + lane]      = c0;
    out_codes[(size_t)row * DDIM + lane + 32] = c1;
    float d0 = c0 - z0, d1 = c1 - z1;
    lossAcc += d0 * d0 + d1 * d1;
    if (lane == 0) out_idx[row] = (float)k;
  }
  #pragma unroll
  for (int off = 1; off < 32; off <<= 1) lossAcc += __shfl_xor(lossAcc, off, 32);
  if (lane == 0) sLoss[wave] = lossAcc;
  __syncthreads();
  if (tid == 0) {                                  // fixed order => deterministic
    float s = 0.f;
    #pragma unroll
    for (int w = 0; w < WAVES; ++w) s += sLoss[w];
    partials[blockIdx.x] = s;
  }
}

// --------------------------------------------------------------------------
// Kernel 3: deterministic reduction of 1024 block partials -> mean
// --------------------------------------------------------------------------
__global__ void vq_finish(const float* __restrict__ partials,
                          float* __restrict__ outLoss) {
  const int lane = threadIdx.x;                    // 32 threads
  float s = 0.f;
  #pragma unroll 1
  for (int i = 0; i < 32; ++i) s += partials[lane * 32 + i];
  #pragma unroll
  for (int off = 1; off < 32; off <<= 1) s += __shfl_xor(s, off, 32);
  if (lane == 0) *outLoss = s * (1.0f / 8388608.0f);   // mean over 16*1024*512
}

// --------------------------------------------------------------------------
extern "C" void kernel_launch(void* const* d_in, const int* in_sizes, int n_in,
                              void* d_out, int out_size, void* d_ws, size_t ws_size,
                              hipStream_t stream) {
  const float* z  = (const float*)d_in[0];     // 8388608 f32
  const float* cb = (const float*)d_in[1];     // 131072 f32 (2048 x 64)

  // workspace: [bf16 codebook 256KB][csq 8KB][partials 4KB]
  unsigned short* cb16 = (unsigned short*)d_ws;
  float* csq      = (float*)((char*)d_ws + (size_t)KCODES * DDIM * 2);
  float* partials = (float*)((char*)d_ws + (size_t)KCODES * DDIM * 2 + KCODES * 4);

  float* out_codes = (float*)d_out;                         // 8388608
  float* out_idx   = out_codes + (size_t)NROWS * DDIM;      // 131072
  float* out_loss  = out_idx + NROWS;                       // 1

  vq_prep  <<<KCODES / 256,            256, 0, stream>>>(cb, cb16, csq);
  vq_main  <<<NROWS / ROWS_PER_BLOCK,  256, 0, stream>>>(z, cb, cb16, csq,
                                                         out_codes, out_idx, partials);
  vq_finish<<<1,                        32, 0, stream>>>(partials, out_loss);
}
